// SERESkippedQwen3MoeSparseMoeBlock_89309549953420
// MI455X (gfx1250) — matrix operand order repaired and checked by reading, not compiled, not run on hardware
//
#include <hip/hip_runtime.h>
#include <hip/hip_bf16.h>

// ---------------------------------------------------------------------------
// MoE (SERESkippedQwen3MoeSparseMoeBlock) for MI455X / gfx1250.
// T=8192 tokens, H=2048, E=32, D_FF=768, TOP_K=8 (4 primary + 4 rerouted).
// Sparse dispatch: router -> per-expert token lists -> bf16 WMMA GEMMs ->
// weighted atomic scatter-add. Memory-bound (~740MB weights+act @ 23.3TB/s).
// ---------------------------------------------------------------------------

#define MOE_E      32
#define MOE_H      2048
#define MOE_DFF    768
#define MOE_T      8192
#define MOE_TOPK   8
#define MOE_SEL    4
#define MOE_THRESH 0.5f
#define TILE_M     16          // tokens per expert tile (one WMMA M tile)

typedef __attribute__((ext_vector_type(16))) __bf16 v16bf;
typedef __attribute__((ext_vector_type(8)))  float  v8f;

union Bf16Frag {
    v16bf  v;
    __bf16 e[16];
    uint4  q[2];
};

__device__ __forceinline__ v8f wmma_bf16(v16bf a, v16bf b, v8f c) {
    // emits v_wmma_f32_16x16x32_bf16
    return __builtin_amdgcn_wmma_f32_16x16x32_bf16(
        /*neg_a=*/false, a, /*neg_b=*/false, b,
        /*c_mod=*/(short)0, c, /*reuse_a=*/false, /*reuse_b=*/false);
}

// Pack two floats into two bf16 (native converts; RNE).
__device__ __forceinline__ unsigned int pk_bf2(float a, float b) {
    union { __bf16 h[2]; unsigned int u; } z;
    z.h[0] = (__bf16)a;
    z.h[1] = (__bf16)b;
    return z.u;
}

// Load 16 consecutive f32 and convert to a bf16 B-fragment slice.
// B k order used here (matches A): lane holds column (lane&15), K = (lane>>4)*16 + 0..15
// -> 16 *contiguous* K elements of one weight row (row-major weights).
__device__ __forceinline__ void load_b_f32(const float* __restrict__ p, Bf16Frag& b) {
#pragma unroll
    for (int i = 0; i < 16; i += 4) {
        float4 t = *(const float4*)(p + i);
        b.e[i + 0] = (__bf16)t.x;
        b.e[i + 1] = (__bf16)t.y;
        b.e[i + 2] = (__bf16)t.z;
        b.e[i + 3] = (__bf16)t.w;
    }
}

// ---------------------------------------------------------------------------
// Kernel 0: zero output accumulator and per-expert counters
// ---------------------------------------------------------------------------
__global__ void moe_zero_kernel(float* __restrict__ y, size_t n, int* __restrict__ counts) {
    size_t i = (size_t)blockIdx.x * blockDim.x + threadIdx.x;
    size_t stride = (size_t)gridDim.x * blockDim.x;
    for (; i < n; i += stride) y[i] = 0.0f;
    if (blockIdx.x == 0 && threadIdx.x < MOE_E) counts[threadIdx.x] = 0;
}

// ---------------------------------------------------------------------------
// Kernel 1: router. One wave32 per token; lane == expert.
// logits -> softmax -> top-8 -> renorm -> similarity reroute -> combine row
// + per-expert token list build.
// ---------------------------------------------------------------------------
__global__ void moe_router_kernel(const float* __restrict__ x,
                                  const float* __restrict__ gw,
                                  const float* __restrict__ sim,
                                  float* __restrict__ combine,
                                  int*   __restrict__ counts,
                                  int*   __restrict__ tlist) {
    const int lane = threadIdx.x & 31;
    const int wave = threadIdx.x >> 5;
    const int tok  = blockIdx.x * (blockDim.x >> 5) + wave;
    if (tok >= MOE_T) return;

    // ---- logits: each lane computes one expert's dot product over H ----
    const float* xr = x  + (size_t)tok  * MOE_H;
    const float* gr = gw + (size_t)lane * MOE_H;
    float acc = 0.0f;
#pragma unroll 4
    for (int k = 0; k < MOE_H; k += 4) {
        float4 xv = *(const float4*)(xr + k);
        float4 gv = *(const float4*)(gr + k);
        acc = fmaf(xv.x, gv.x, acc);
        acc = fmaf(xv.y, gv.y, acc);
        acc = fmaf(xv.z, gv.z, acc);
        acc = fmaf(xv.w, gv.w, acc);
    }

    // ---- softmax across the 32 lanes (wave32 shuffles) ----
    float m = acc;
#pragma unroll
    for (int off = 16; off >= 1; off >>= 1) {
        float o = __shfl_xor(m, off, 32);
        m = fmaxf(m, o);
    }
    float p = __expf(acc - m);
    float s = p;
#pragma unroll
    for (int off = 16; off >= 1; off >>= 1) s += __shfl_xor(s, off, 32);
    float prob = p / s;

    // ---- top-8 (descending; ties -> lower expert index, like jax top_k) ----
    float v = prob;
    float topv[MOE_TOPK];
    int   topi[MOE_TOPK];
#pragma unroll
    for (int j = 0; j < MOE_TOPK; ++j) {
        float bv = v;
        int   bi = lane;
#pragma unroll
        for (int off = 16; off >= 1; off >>= 1) {
            float ov = __shfl_xor(bv, off, 32);
            int   oi = __shfl_xor(bi, off, 32);
            if (ov > bv || (ov == bv && oi < bi)) { bv = ov; bi = oi; }
        }
        topv[j] = bv;            // broadcast: identical in every lane
        topi[j] = bi;
        if (lane == bi) v = -1.0f;   // remove winner
    }

    // ---- renormalize over the original top-8 values ----
    float sum8 = 0.0f;
#pragma unroll
    for (int j = 0; j < MOE_TOPK; ++j) sum8 += topv[j];
    sum8 = fmaxf(sum8, 1e-12f);
    float w[MOE_TOPK];
#pragma unroll
    for (int j = 0; j < MOE_TOPK; ++j) w[j] = topv[j] / sum8;

    // ---- similarity rerouting; lane e accumulates combine[tok][e] ----
    float myc = 0.0f;
#pragma unroll
    for (int j = 0; j < MOE_SEL; ++j)
        if (topi[j] == lane) myc += w[j];
#pragma unroll
    for (int j = MOE_SEL; j < MOE_TOPK; ++j) {
        int   se = topi[j];
        float bs = -1.0f;
        int   bp = 0;
#pragma unroll
        for (int i = 0; i < MOE_SEL; ++i) {
            int   pe = topi[i];
            float sv = (se == pe) ? 1.0f : sim[se * MOE_E + pe];  // diag forced to 1
            if (sv > bs) { bs = sv; bp = pe; }                    // first-max (argmax)
        }
        int ne = (bs < MOE_THRESH) ? se : bp;
        if (ne == lane) myc += w[j];
    }
    combine[(size_t)tok * MOE_E + lane] = myc;

    if (myc != 0.0f) {
        int pos = atomicAdd(&counts[lane], 1);
        tlist[(size_t)lane * MOE_T + pos] = tok;
    }
}

// ---------------------------------------------------------------------------
// Kernel 2: per-expert fused gate_up -> silu*u -> down -> weighted scatter.
// Block = 256 threads (8 waves), one 16-token tile of one expert's list.
// Dynamic LDS: sTok(64B) sCw(64B) sX(16x2048 bf16 = 64KB) sH(16x768 bf16 = 24KB)
// ---------------------------------------------------------------------------
__global__ void moe_expert_kernel(const float* __restrict__ x,
                                  const float* __restrict__ gup,   // [E][2*DFF][H]
                                  const float* __restrict__ dwn,   // [E][H][DFF]
                                  const float* __restrict__ combine,
                                  const int*   __restrict__ counts,
                                  const int*   __restrict__ tlist,
                                  float*       __restrict__ y) {
    extern __shared__ unsigned char smem[];
    int*    sTok = (int*)smem;                                   // 64 B
    float*  sCw  = (float*)(smem + 64);                          // 64 B
    __bf16* sX   = (__bf16*)(smem + 128);                        // 64 KB
    __bf16* sH   = (__bf16*)(smem + 128 + TILE_M * MOE_H * 2);   // 24 KB

    const int e    = blockIdx.y;
    const int cnt  = counts[e];
    const int row0 = blockIdx.x * TILE_M;
    if (row0 >= cnt) return;                       // uniform early exit

    const int tid  = threadIdx.x;
    const int lane = tid & 31;
    const int wave = tid >> 5;                     // 0..7
    const int col  = lane & 15;                    // WMMA N / M row index
    const int hi   = lane >> 4;                    // lane half (0/1)
    const int koff = hi * 16;                      // A-fragment K offset (halves)

    if (tid < TILE_M) {
        int r   = row0 + tid;
        int tk  = (r < cnt) ? tlist[(size_t)e * MOE_T + r]
                            : tlist[(size_t)e * MOE_T + row0];   // safe pad token
        float c = (r < cnt) ? combine[(size_t)tk * MOE_E + e] : 0.0f;
        sTok[tid] = tk;
        sCw[tid]  = c;
    }
    __syncthreads();

    // ---- stage X tile into LDS as bf16 (16 rows x 2048) ----
    {
        int r  = tid >> 4;                 // 0..15
        int c0 = (tid & 15) * 128;         // 128 floats per thread
        const float* src = x + (size_t)sTok[r] * MOE_H + c0;
        __bf16* dst = &sX[r * MOE_H + c0];
#pragma unroll 4
        for (int i = 0; i < 128; i += 4) {
            float4 t = *(const float4*)(src + i);
            unsigned long long pk =
                  (unsigned long long)pk_bf2(t.x, t.y)
                | ((unsigned long long)pk_bf2(t.z, t.w) << 32);
            *(unsigned long long*)(dst + i) = pk;
        }
    }
    __syncthreads();

    // =====================  gate_up + silu*u  =====================
    // gu[m][n] over n in [0,1536); pair column tile j (g) with j+DFF (u).
    const float* wg = gup + (size_t)e * (2 * MOE_DFF) * MOE_H;
    const __bf16* aRow = &sX[col * MOE_H];              // A row = lane&15

    for (int jt = wave; jt < MOE_DFF / 16; jt += 8) {   // 48 tiles, 6 per wave
        const float* bg = wg + (size_t)(jt * 16 + col) * MOE_H + hi * 16;
        const float* bu = wg + (size_t)(MOE_DFF + jt * 16 + col) * MOE_H + hi * 16;
        v8f accg = {};
        v8f accu = {};
        for (int kk = 0; kk < MOE_H; kk += 32) {
            Bf16Frag a, fg, fu;
            a.q[0] = *(const uint4*)(aRow + kk + koff);        // K 0..7 / 16..23
            a.q[1] = *(const uint4*)(aRow + kk + 8 + koff);    // K 8..15 / 24..31
            __builtin_prefetch(bg + kk + 128, 0, 0);           // global_prefetch_b8
            load_b_f32(bg + kk, fg);
            load_b_f32(bu + kk, fu);
            accg = wmma_bf16(a.v, fg.v, accg);
            accu = wmma_bf16(a.v, fu.v, accu);
        }
        // h = silu(g) * u  ->  LDS bf16 (C/D layout: m = r + 8*hi, n = tile+col)
        // silu via fast v_rcp_f32 instead of the IEEE div expansion.
#pragma unroll
        for (int r = 0; r < 8; ++r) {
            float g = accg[r];
            float u = accu[r];
            float h = g * u * __builtin_amdgcn_rcpf(1.0f + __expf(-g));
            int mrow = r + hi * 8;
            sH[mrow * MOE_DFF + jt * 16 + col] = (__bf16)h;
        }
    }
    __syncthreads();

    // =====================  down projection + scatter  =====================
    const float* wd = dwn + (size_t)e * MOE_H * MOE_DFF;
    const __bf16* hRow = &sH[col * MOE_DFF];

    for (int nt = wave; nt < MOE_H / 16; nt += 8) {     // 128 tiles, 16 per wave
        const float* bb = wd + (size_t)(nt * 16 + col) * MOE_DFF + hi * 16;
        v8f acc = {};
        for (int kk = 0; kk < MOE_DFF; kk += 32) {
            Bf16Frag a, b;
            a.q[0] = *(const uint4*)(hRow + kk + koff);
            a.q[1] = *(const uint4*)(hRow + kk + 8 + koff);
            __builtin_prefetch(bb + kk + 128, 0, 0);
            load_b_f32(bb + kk, b);
            acc = wmma_bf16(a.v, b.v, acc);
        }
#pragma unroll
        for (int r = 0; r < 8; ++r) {
            int   mrow = r + hi * 8;
            float val  = acc[r] * sCw[mrow];            // 0 for pad rows
            atomicAdd(&y[(size_t)sTok[mrow] * MOE_H + nt * 16 + col], val);
        }
    }
}

// ---------------------------------------------------------------------------
// Launch
// ---------------------------------------------------------------------------
extern "C" void kernel_launch(void* const* d_in, const int* in_sizes, int n_in,
                              void* d_out, int out_size, void* d_ws, size_t ws_size,
                              hipStream_t stream) {
    const float* x   = (const float*)d_in[0];   // (4,2048,2048)
    const float* gw  = (const float*)d_in[1];   // (32,2048)
    const float* gup = (const float*)d_in[2];   // (32,1536,2048)
    const float* dwn = (const float*)d_in[3];   // (32,2048,768)
    const float* sim = (const float*)d_in[4];   // (32,32)
    float* y = (float*)d_out;                   // (4,2048,2048)

    // workspace carve (~2.1 MB)
    float* combine = (float*)d_ws;                               // T*E f32
    int*   counts  = (int*)(combine + (size_t)MOE_T * MOE_E);    // E ints
    int*   tlist   = counts + MOE_E;                             // E*T ints
    (void)in_sizes; (void)n_in; (void)out_size; (void)ws_size;

    const size_t yN = (size_t)MOE_T * MOE_H;
    moe_zero_kernel<<<2048, 256, 0, stream>>>(y, yN, counts);

    // 8 waves/block, one token per wave
    moe_router_kernel<<<MOE_T / 8, 256, 0, stream>>>(x, gw, sim, combine, counts, tlist);

    // 512 tiles x 32 experts; blocks past an expert's count exit immediately
    const size_t lds = 128 + (size_t)TILE_M * MOE_H * 2 + (size_t)TILE_M * MOE_DFF * 2; // 90240 B
    dim3 grid(MOE_T / TILE_M, MOE_E, 1);
    moe_expert_kernel<<<grid, 256, lds, stream>>>(x, gup, dwn, combine, counts, tlist, y);
}
